// MERFFGNN_52475910423290
// MI455X (gfx1250) — matrix operand order repaired and checked by reading, not compiled, not run on hardware
//
#include <hip/hip_runtime.h>
#include <hip/hip_bf16.h>

// ---------------- problem constants ----------------
constexpr int FRAMES  = 4;
constexpr int ATOMS   = 1024;
constexpr int NA      = FRAMES * ATOMS;   // 4096 atoms
constexpr int EDGES   = 65536;
constexpr int HID     = 1024;
constexpr int NG      = 50;
constexpr int NGPAD   = 64;               // RBF padded to K multiple of 32
constexpr int D_RFF   = 10000;
constexpr int QLD     = 10016;            // D_RFF padded to multiple of 32
constexpr float CUTOFF = 6.0f;
constexpr float SIGMA  = 10.0f;
constexpr float PI_F   = 3.14159265358979323846f;
constexpr float LN2_F  = 0.69314718055994531f;

// ---------------- WMMA types ----------------
typedef _Float16 v16h __attribute__((ext_vector_type(16)));
typedef _Float16 v8h  __attribute__((ext_vector_type(8)));
typedef float    v8f  __attribute__((ext_vector_type(8)));
typedef int      v4i  __attribute__((ext_vector_type(4)));

#define LDSS 48   // LDS row stride in halves (96B: keeps b128 alignment, skews banks)

// CDNA5 async global->LDS copy (raw bytes, tracked by ASYNCcnt)
#if defined(__has_builtin)
#  if __has_builtin(__builtin_amdgcn_global_load_async_to_lds_b128) && \
      __has_builtin(__builtin_amdgcn_s_wait_asynccnt)
#    define USE_ASYNC_LDS 1
#  endif
#endif

// ======================================================================
// f32-activation x f16-weight WMMA GEMM:  C[M,N] = epilogue(alpha * A @ B^T)
//   A: [M,K] f32 row major (lda), zero padded up to K (K % 32 == 0)
//   B: [N,K] f16 row major (ldb), K fully valid (weights pre-padded)
//   mode 0: x = alpha*(acc + bias)
//   mode 1: x = ssp(acc + bias)                    (shifted softplus)
//   mode 2: x = acc + bias + aux[m,n]              (residual add)
//   mode 3: x = acc * (1 - 0.5*exp(-aux[m,n]))     (mul by sigmoid, aux = ssp act)
// Block: 128x128x32 tile, 256 threads = 8 wave32 waves (2x4), wave tile 64x32.
// LDS double-buffered; B tiles staged via async global->LDS b128 copies.
// ======================================================================
__launch_bounds__(256)
__global__ void gemm_wmma_f16(const float* __restrict__ A, long lda,
                              const _Float16* __restrict__ B, long ldb,
                              float* __restrict__ C, long ldc,
                              int M, int N, int K,
                              float alpha,
                              const float* __restrict__ bias,
                              const float* __restrict__ aux, int mode)
{
    __shared__ __align__(16) _Float16 sA[2][128 * LDSS];
    __shared__ __align__(16) _Float16 sB[2][128 * LDSS];

    const int t    = threadIdx.x;
    const int wave = t >> 5;
    const int lane = t & 31;
    const int wy   = wave >> 2;            // 0..1  (rows of waves)
    const int wx   = wave & 3;             // 0..3  (cols of waves)
    const int mwave = wy * 64;             // wave tile: 64 x 32
    const int nwave = wx * 32;
    const int blockM = blockIdx.y * 128;
    const int blockN = blockIdx.x * 128;

    const bool fullM = (blockM + 128 <= M);
    const bool fullN = (blockN + 128 <= N);

    const int lrow  = lane & 15;
    const int khalf = (lane >> 4) << 3;    // A frag: halves 0 or 8
    const int bk0   = (lane >> 4) << 3;    // B frag: halves 0 or 8

    // ---- A stager: b128 f32 loads (phase 1), cvt + ds stores (phase 2) ----
    auto stageA = [&](int k0, int buf) {
        if (fullM) {
            float4 va[4];
#pragma unroll
            for (int i = 0; i < 4; ++i) {
                int idx = i * 256 + t;                 // 1024 float4s
                int r = idx >> 3, c4 = (idx & 7) << 2;
                va[i] = *(const float4*)(A + (size_t)(blockM + r) * lda + k0 + c4);
            }
#pragma unroll
            for (int i = 0; i < 4; ++i) {
                int idx = i * 256 + t;
                int r = idx >> 3, c4 = (idx & 7) << 2;
                _Float16* dst = &sA[buf][r * LDSS + c4];
                dst[0] = (_Float16)va[i].x; dst[1] = (_Float16)va[i].y;
                dst[2] = (_Float16)va[i].z; dst[3] = (_Float16)va[i].w;
            }
        } else {
#pragma unroll
            for (int i = 0; i < 4; ++i) {
                int idx = i * 256 + t;
                int r = idx >> 3, c4 = (idx & 7) << 2;
                float4 v = make_float4(0.f, 0.f, 0.f, 0.f);
                if (blockM + r < M)
                    v = *(const float4*)(A + (size_t)(blockM + r) * lda + k0 + c4);
                _Float16* dst = &sA[buf][r * LDSS + c4];
                dst[0] = (_Float16)v.x; dst[1] = (_Float16)v.y;
                dst[2] = (_Float16)v.z; dst[3] = (_Float16)v.w;
            }
        }
    };

    // ---- B stager: raw f16 16B chunks; async DMA to LDS on CDNA5 ----
    auto stageB = [&](int k0, int buf) {
        if (fullN) {
#pragma unroll
            for (int i = 0; i < 2; ++i) {
                int idx = i * 256 + t;                 // 512 x 16B chunks
                int n = idx >> 2, c8 = (idx & 3) << 3; // half-offset 0/8/16/24
                const _Float16* gsrc = B + (size_t)(blockN + n) * ldb + k0 + c8;
                _Float16* ldst = &sB[buf][n * LDSS + c8];
#ifdef USE_ASYNC_LDS
                __builtin_amdgcn_global_load_async_to_lds_b128(
                    (v4i*)gsrc, (v4i*)ldst, 0, 0);
#else
                *(v8h*)ldst = *(const v8h*)gsrc;
#endif
            }
        } else {
            // N tail (q GEMM): guarded scalar f16 copies
#pragma unroll
            for (int i = 0; i < 16; ++i) {
                int idx = i * 256 + t;                 // 4096 halves
                int n = idx & 127, k = idx >> 7;
                int gn = blockN + n;
                _Float16 v = (_Float16)0.f;
                if (gn < N) v = B[(size_t)gn * ldb + k0 + k];
                sB[buf][n * LDSS + k] = v;
            }
        }
    };

    v8f acc[4][2] = {};

    stageA(0, 0);
    stageB(0, 0);
#ifdef USE_ASYNC_LDS
    __builtin_amdgcn_s_wait_asynccnt(0);
#endif
    __syncthreads();

    int buf = 0;
    for (int k0 = 0; k0 < K; k0 += 32) {
        const int nk = k0 + 32;
        if (nk < K) {
            stageA(nk, buf ^ 1);     // VMEM + async DMA issued before the WMMA block
            stageB(nk, buf ^ 1);
            if (nk + 32 < K && fullM) {
                int gm = blockM + (t >> 1);
                __builtin_prefetch(A + (size_t)gm * lda + nk + 32 + ((t & 1) << 4), 0, 0);
            }
        }

        // ---- B fragments (lanes<16: K 0-7/16-23, lanes>=16: K 8-15/24-31)
        v16h bfrag[2];
#pragma unroll
        for (int nt = 0; nt < 2; ++nt) {
            const _Float16* p = &sB[buf][(nwave + nt * 16 + lrow) * LDSS + bk0];
            v8h lo = *(const v8h*)p;
            v8h hi = *(const v8h*)(p + 16);
#pragma unroll
            for (int j = 0; j < 8; ++j) { bfrag[nt][j] = lo[j]; bfrag[nt][8 + j] = hi[j]; }
        }
        // ---- A fragments (lanes<16: K 0-7/16-23, lanes>=16: K 8-15/24-31) + 8x WMMA
#pragma unroll
        for (int mt = 0; mt < 4; ++mt) {
            const _Float16* p = &sA[buf][(mwave + mt * 16 + lrow) * LDSS + khalf];
            v8h lo = *(const v8h*)p;
            v8h hi = *(const v8h*)(p + 16);
            v16h afrag;
#pragma unroll
            for (int j = 0; j < 8; ++j) { afrag[j] = lo[j]; afrag[8 + j] = hi[j]; }
#pragma unroll
            for (int nt = 0; nt < 2; ++nt)
                acc[mt][nt] = __builtin_amdgcn_wmma_f32_16x16x32_f16(
                    false, afrag, false, bfrag[nt], (short)0, acc[mt][nt], false, false);
        }
#ifdef USE_ASYNC_LDS
        __builtin_amdgcn_s_wait_asynccnt(0);   // next-buffer DMA done (after compute)
#endif
        __syncthreads();
        buf ^= 1;
    }

    // ---- epilogue + store (C/D layout: VGPR j -> M=j / M=8+j, lane&15 -> N) ----
    const int rlo = (lane >> 4) << 3;      // 0 or 8
    const int cn  = lane & 15;
#pragma unroll
    for (int mt = 0; mt < 4; ++mt)
#pragma unroll
        for (int nt = 0; nt < 2; ++nt)
#pragma unroll
            for (int j = 0; j < 8; ++j) {
                int gm = blockM + mwave + mt * 16 + rlo + j;
                int gn = blockN + nwave + nt * 16 + cn;
                if (gm < M && gn < N) {
                    float x = acc[mt][nt][j];
                    if (bias) x += bias[gn];
                    x *= alpha;
                    if (mode == 1) {
                        x = ((x > 20.f) ? x : log1pf(__expf(x))) - LN2_F;
                    } else if (mode == 2) {
                        x += aux[(size_t)gm * ldc + gn];
                    } else if (mode == 3) {
                        x *= (1.0f - 0.5f * __expf(-aux[(size_t)gm * ldc + gn]));
                    }
                    C[(size_t)gm * ldc + gn] = x;
                }
            }
}

// ======================================================================
// One-time weight conversion: f32 [R,C] -> f16 [R,Cpad] (zero-padded)
// ======================================================================
__global__ void cvt_f16_pad(const float* __restrict__ src, _Float16* __restrict__ dst,
                            int R, int C, int Cpad)
{
    int idx = blockIdx.x * blockDim.x + threadIdx.x;
    if (idx >= R * Cpad) return;
    int r = idx / Cpad, c = idx % Cpad;
    dst[idx] = (c < C) ? (_Float16)src[(size_t)r * C + c] : (_Float16)0.f;
}

// f32 [R,C] -> transposed f16 [Cpad, Rpad]: dst[c,r] = src[r,c] (zero-padded)
__global__ void cvt_f16_T(const float* __restrict__ src, _Float16* __restrict__ dst,
                          int R, int C, int Rpad, int Cpad)
{
    int idx = blockIdx.x * blockDim.x + threadIdx.x;
    if (idx >= Cpad * Rpad) return;
    int c = idx / Rpad, r = idx % Rpad;
    dst[idx] = (c < C && r < R) ? (_Float16)src[(size_t)r * C + c] : (_Float16)0.f;
}

// ======================================================================
// Edge geometry: dist, cosine cutoff, padded Gaussian RBF
// ======================================================================
__global__ void edge_geom(const float* __restrict__ pos, const int* __restrict__ ei,
                          float* __restrict__ dist, float* __restrict__ ccut,
                          float* __restrict__ rbf)
{
    int e = blockIdx.x * blockDim.x + threadIdx.x;
    if (e >= EDGES) return;
    int r = ei[e], c = ei[EDGES + e];
    float dx = pos[r * 3 + 0] - pos[c * 3 + 0];
    float dy = pos[r * 3 + 1] - pos[c * 3 + 1];
    float dz = pos[r * 3 + 2] - pos[c * 3 + 2];
    float d  = sqrtf(dx * dx + dy * dy + dz * dz + 1e-12f);
    dist[e] = d;
    ccut[e] = (d < CUTOFF) ? 0.5f * (__cosf(d * PI_F / CUTOFF) + 1.0f) : 0.0f;
    const float step  = CUTOFF / (float)(NG - 1);
    const float coeff = -0.5f / (step * step);
    float* re = rbf + (size_t)e * NGPAD;
#pragma unroll 4
    for (int g = 0; g < NGPAD; ++g) {
        float dm = d - (float)g * step;
        re[g] = (g < NG) ? __expf(coeff * dm * dm) : 0.0f;
    }
}

// embedding gather x0 = emb[z]
__global__ void build_x0(const float* __restrict__ emb, const int* __restrict__ z,
                         float* __restrict__ x0)
{
    int idx = blockIdx.x * blockDim.x + threadIdx.x;   // over NA*HID/4
    if (idx >= NA * (HID / 4)) return;
    int n = idx / (HID / 4), c = idx % (HID / 4);
    const float4* src = (const float4*)(emb + (size_t)z[n] * HID);
    ((float4*)(x0 + (size_t)n * HID))[c] = src[c];
}

// agg[row[e]] += y[col[e]] * Wpre[e] * C[e]
__global__ void scatter_agg(const float* __restrict__ y, const float* __restrict__ wpre,
                            const float* __restrict__ ccut, const int* __restrict__ ei,
                            float* __restrict__ agg)
{
    int e = blockIdx.x, t = threadIdx.x;
    int r = ei[e], c = ei[EDGES + e];
    float ce = ccut[e];
    const float4* yv = (const float4*)(y    + (size_t)c * HID);
    const float4* wv = (const float4*)(wpre + (size_t)e * HID);
    float* an = agg + (size_t)r * HID;
    for (int i = t; i < HID / 4; i += 256) {
        float4 a = yv[i], w = wv[i];
        atomicAdd(&an[i * 4 + 0], a.x * w.x * ce);
        atomicAdd(&an[i * 4 + 1], a.y * w.y * ce);
        atomicAdd(&an[i * 4 + 2], a.z * w.z * ce);
        atomicAdd(&an[i * 4 + 3], a.w * w.w * ce);
    }
}

// energy: E_f = g/ATOMS * sum_a sum_D cos(q)*wc + sin(q)*ws
__global__ void energy_red(const float* __restrict__ q, const float* __restrict__ wout,
                           float* __restrict__ out)
{
    __shared__ float red[256];
    int f = blockIdx.y;
    int D = blockIdx.x * 256 + threadIdx.x;
    float s = 0.f;
    if (D < D_RFF) {
        float wc = wout[D], ws = wout[D_RFF + D];
        const float* qp = q + (size_t)(f * ATOMS) * QLD + D;
        for (int a = 0; a < ATOMS; ++a) {
            float qv = qp[(size_t)a * QLD];
            s += __cosf(qv) * wc + __sinf(qv) * ws;
        }
    }
    red[threadIdx.x] = s;
    __syncthreads();
    for (int o = 128; o > 0; o >>= 1) {
        if (threadIdx.x < o) red[threadIdx.x] += red[threadIdx.x + o];
        __syncthreads();
    }
    if (threadIdx.x == 0) {
        float g = sqrtf(2.0f / (float)D_RFF);
        atomicAdd(&out[f], red[0] * (g / (float)ATOMS));
    }
}

// in-place q -> dq (also zeroes the K padding columns)
__global__ void dq_inplace(float* __restrict__ q, const float* __restrict__ wout)
{
    int m = blockIdx.y;
    int D = blockIdx.x * 256 + threadIdx.x;
    if (D >= QLD) return;
    float* p = q + (size_t)m * QLD + D;
    if (D < D_RFF) {
        float g  = sqrtf(2.0f / (float)D_RFF);
        float qv = *p;
        *p = (g / (float)ATOMS) * (-__sinf(qv) * wout[D] + __cosf(qv) * wout[D_RFF + D]);
    } else {
        *p = 0.0f;
    }
}

// per-edge backward through the scatter: Wpre <- dWpre, and dCdot[e]
__global__ void edge_back(const float* __restrict__ dagg, const float* __restrict__ y,
                          float* __restrict__ wpre, const float* __restrict__ ccut,
                          const int* __restrict__ ei, float* __restrict__ dcdot)
{
    __shared__ float red[256];
    int e = blockIdx.x, t = threadIdx.x;
    int r = ei[e], c = ei[EDGES + e];
    float ce = ccut[e];
    const float* dr = dagg + (size_t)r * HID;
    const float* yc = y    + (size_t)c * HID;
    float* wp = wpre + (size_t)e * HID;
    float s = 0.f;
    for (int i = t; i < HID; i += 256) {
        float g = dr[i] * yc[i];
        s += g * wp[i];
        wp[i] = g * ce;
    }
    red[t] = s;
    __syncthreads();
    for (int o = 128; o > 0; o >>= 1) {
        if (t < o) red[t] += red[t + o];
        __syncthreads();
    }
    if (t == 0) dcdot[e] = red[0];
}

// ddist from drbf + dC, then accumulate position gradients
__global__ void dist_back(const float* __restrict__ pos, const int* __restrict__ ei,
                          const float* __restrict__ dist, const float* __restrict__ drbf,
                          const float* __restrict__ dcdot, float* __restrict__ gpos)
{
    int e = blockIdx.x * blockDim.x + threadIdx.x;
    if (e >= EDGES) return;
    float d = dist[e];
    const float step  = CUTOFF / (float)(NG - 1);
    const float coeff = -0.5f / (step * step);
    const float* dre = drbf + (size_t)e * NGPAD;
    float dd = 0.f;
#pragma unroll 5
    for (int g = 0; g < NG; ++g) {
        float dm = d - (float)g * step;
        dd += dre[g] * __expf(coeff * dm * dm) * 2.0f * coeff * dm;
    }
    if (d < CUTOFF)
        dd += dcdot[e] * (-0.5f * __sinf(d * PI_F / CUTOFF) * (PI_F / CUTOFF));
    int r = ei[e], c = ei[EDGES + e];
    float dx = pos[r * 3 + 0] - pos[c * 3 + 0];
    float dy = pos[r * 3 + 1] - pos[c * 3 + 1];
    float dz = pos[r * 3 + 2] - pos[c * 3 + 2];
    float inv = dd / d;
    atomicAdd(&gpos[r * 3 + 0],  inv * dx);
    atomicAdd(&gpos[r * 3 + 1],  inv * dy);
    atomicAdd(&gpos[r * 3 + 2],  inv * dz);
    atomicAdd(&gpos[c * 3 + 0], -inv * dx);
    atomicAdd(&gpos[c * 3 + 1], -inv * dy);
    atomicAdd(&gpos[c * 3 + 2], -inv * dz);
}

__global__ void finalize_forces(const float* __restrict__ gpos, float* __restrict__ out)
{
    int idx = blockIdx.x * blockDim.x + threadIdx.x;
    if (idx < NA * 3) out[FRAMES + idx] = -gpos[idx];
}

// ======================================================================
extern "C" void kernel_launch(void* const* d_in, const int* in_sizes, int n_in,
                              void* d_out, int out_size, void* d_ws, size_t ws_size,
                              hipStream_t stream)
{
    (void)in_sizes; (void)n_in; (void)out_size; (void)ws_size;

    const float* pos   = (const float*)d_in[0];
    const float* emb   = (const float*)d_in[1];
    const float* fw1   = (const float*)d_in[2];
    const float* fb1   = (const float*)d_in[3];
    const float* fw2   = (const float*)d_in[4];
    const float* fb2   = (const float*)d_in[5];
    const float* l1w   = (const float*)d_in[6];
    const float* l2w   = (const float*)d_in[7];
    const float* l2b   = (const float*)d_in[8];
    const float* l3w   = (const float*)d_in[9];
    const float* l3b   = (const float*)d_in[10];
    const float* w_rff = (const float*)d_in[11];
    const float* w_out = (const float*)d_in[12];
    const int*   zat   = (const int*)d_in[13];
    const int*   ei    = (const int*)d_in[14];
    float* out = (float*)d_out;
    float* ws  = (float*)d_ws;

    // ---- f32 workspace ----
    size_t o = 0;
    float* dist  = ws + o; o += EDGES;
    float* ccut  = ws + o; o += EDGES;
    float* dcdot = ws + o; o += EDGES;
    float* rbf   = ws + o; o += (size_t)EDGES * NGPAD;   // reused as drbf
    float* t1    = ws + o; o += (size_t)EDGES * HID;     // reused as dt1pre
    float* wpre  = ws + o; o += (size_t)EDGES * HID;     // reused as dWpre
    float* x0    = ws + o; o += (size_t)NA * HID;        // reused as dagg
    float* ybuf  = ws + o; o += (size_t)NA * HID;
    float* agg   = ws + o; o += (size_t)NA * HID;        // reused as dv
    float* sspv  = ws + o; o += (size_t)NA * HID;
    float* hbuf  = ws + o; o += (size_t)NA * HID;        // reused as dh
    float* qbuf  = ws + o; o += (size_t)NA * QLD;        // reused as dq (in place)
    float* gpos  = ws + o; o += (size_t)NA * 3;

    // ---- f16 weight workspace (converted once per launch) ----
    _Float16* hws = (_Float16*)(ws + o);
    size_t ho = 0;
    _Float16* fw1h  = hws + ho; ho += (size_t)HID * NGPAD;      // [1024,64]
    _Float16* fw2h  = hws + ho; ho += (size_t)HID * HID;
    _Float16* l1wh  = hws + ho; ho += (size_t)HID * HID;
    _Float16* l2wh  = hws + ho; ho += (size_t)HID * HID;
    _Float16* l3wh  = hws + ho; ho += (size_t)HID * HID;
    _Float16* wrffh = hws + ho; ho += (size_t)D_RFF * HID;      // [10000,1024]
    _Float16* fw2t  = hws + ho; ho += (size_t)HID * HID;        // fw2^T
    _Float16* l2wt  = hws + ho; ho += (size_t)HID * HID;
    _Float16* l3wt  = hws + ho; ho += (size_t)HID * HID;
    _Float16* fw1t  = hws + ho; ho += (size_t)NGPAD * HID;      // [64,1024] rows>=50 zero
    _Float16* wrfft = hws + ho; ho += (size_t)HID * QLD;        // [1024,10016] cols>=10000 zero

    (void)hipMemsetAsync(agg,  0, (size_t)NA * HID * sizeof(float), stream);
    (void)hipMemsetAsync(gpos, 0, (size_t)NA * 3 * sizeof(float), stream);
    (void)hipMemsetAsync(out,  0, FRAMES * sizeof(float), stream);

    auto cvt = [&](const float* s, _Float16* d, int R, int C, int Cpad) {
        int tot = R * Cpad;
        cvt_f16_pad<<<(tot + 255) / 256, 256, 0, stream>>>(s, d, R, C, Cpad);
    };
    auto cvtT = [&](const float* s, _Float16* d, int R, int C, int Rpad, int Cpad) {
        int tot = Cpad * Rpad;
        cvt_f16_T<<<(tot + 255) / 256, 256, 0, stream>>>(s, d, R, C, Rpad, Cpad);
    };

    // weight conversion (one cheap pass; all buffers K-padded -> no GEMM guards)
    cvt (fw1,   fw1h,  HID,   NG,  NGPAD);
    cvt (fw2,   fw2h,  HID,   HID, HID);
    cvt (l1w,   l1wh,  HID,   HID, HID);
    cvt (l2w,   l2wh,  HID,   HID, HID);
    cvt (l3w,   l3wh,  HID,   HID, HID);
    cvt (w_rff, wrffh, D_RFF, HID, HID);
    cvtT(fw2,   fw2t,  HID,   HID, HID,   HID);
    cvtT(l2w,   l2wt,  HID,   HID, HID,   HID);
    cvtT(l3w,   l3wt,  HID,   HID, HID,   HID);
    cvtT(fw1,   fw1t,  HID,   NG,  HID,   NGPAD);   // [64,1024]
    cvtT(w_rff, wrfft, D_RFF, HID, QLD,   HID);     // [1024,10016]

    auto gemm = [&](const float* A, long lda, const _Float16* B, long ldb,
                    float* C, long ldc, int M, int Nn, int K,
                    float alpha, const float* bias, const float* aux, int mode) {
        dim3 g((Nn + 127) / 128, (M + 127) / 128);
        gemm_wmma_f16<<<g, dim3(256), 0, stream>>>(A, lda, B, ldb, C, ldc,
                                                   M, Nn, K, alpha, bias, aux, mode);
    };

    // ---------------- forward ----------------
    edge_geom<<<EDGES / 256, 256, 0, stream>>>(pos, ei, dist, ccut, rbf);
    build_x0<<<(NA * (HID / 4) + 255) / 256, 256, 0, stream>>>(emb, zat, x0);

    // t1 = ssp(rbf @ fw1^T + fb1)                     [E,H]
    gemm(rbf, NGPAD, fw1h, NGPAD, t1, HID, EDGES, HID, NGPAD, 1.0f, fb1, nullptr, 1);
    // Wpre = t1 @ fw2^T + fb2                         [E,H]
    gemm(t1, HID, fw2h, HID, wpre, HID, EDGES, HID, HID, 1.0f, fb2, nullptr, 0);
    // y = x0 @ l1w^T                                  [N,H]
    gemm(x0, HID, l1wh, HID, ybuf, HID, NA, HID, HID, 1.0f, nullptr, nullptr, 0);
    // agg[row] += y[col] * Wpre * C
    scatter_agg<<<EDGES, 256, 0, stream>>>(ybuf, wpre, ccut, ei, agg);
    // sspv = ssp(agg @ l2w^T + l2b)                   [N,H]
    gemm(agg, HID, l2wh, HID, sspv, HID, NA, HID, HID, 1.0f, l2b, nullptr, 1);
    // h = sspv @ l3w^T + l3b + x0                     [N,H]
    gemm(sspv, HID, l3wh, HID, hbuf, HID, NA, HID, HID, 1.0f, l3b, x0, 2);
    // q = (h @ w_rff^T) / SIGMA                       [N,D_RFF] (ld QLD)
    gemm(hbuf, HID, wrffh, HID, qbuf, QLD, NA, D_RFF, HID, 1.0f / SIGMA,
         nullptr, nullptr, 0);
    // energy
    energy_red<<<dim3((D_RFF + 255) / 256, FRAMES), 256, 0, stream>>>(qbuf, w_out, out);

    // ---------------- backward to pos ----------------
    // q -> dq in place (pads zeroed)
    dq_inplace<<<dim3((QLD + 255) / 256, NA), 256, 0, stream>>>(qbuf, w_out);
    // dh = (dq @ w_rff) / SIGMA = dq @ (w_rff^T)^T    [N,H]
    gemm(qbuf, QLD, wrfft, QLD, hbuf, HID, NA, HID, QLD, 1.0f / SIGMA,
         nullptr, nullptr, 0);
    // dv = (dh @ l3w) * sigmoid-from-sspv             [N,H]
    gemm(hbuf, HID, l3wt, HID, agg, HID, NA, HID, HID, 1.0f, nullptr, sspv, 3);
    // dagg = dv @ l2w                                 [N,H]
    gemm(agg, HID, l2wt, HID, x0, HID, NA, HID, HID, 1.0f, nullptr, nullptr, 0);
    // edge backward: Wpre <- dWpre, dCdot
    edge_back<<<EDGES, 256, 0, stream>>>(x0, ybuf, wpre, ccut, ei, dcdot);
    // dt1pre = (dWpre @ fw2) * sigmoid-from-t1        [E,H]
    gemm(wpre, HID, fw2t, HID, t1, HID, EDGES, HID, HID, 1.0f, nullptr, t1, 3);
    // drbf = dt1pre @ fw1                             [E,64] (cols 50..63 unused)
    gemm(t1, HID, fw1t, HID, rbf, NGPAD, EDGES, NGPAD, HID, 1.0f, nullptr, nullptr, 0);
    // ddist -> gradpos
    dist_back<<<EDGES / 256, 256, 0, stream>>>(pos, ei, dist, rbf, dcdot, gpos);
    // forces = -gradpos
    finalize_forces<<<(NA * 3 + 255) / 256, 256, 0, stream>>>(gpos, out);
}
